// conv_layer_13726715478121
// MI455X (gfx1250) — matrix operand order repaired and checked by reading, not compiled, not run on hardware
//
#include <hip/hip_runtime.h>

typedef __attribute__((ext_vector_type(16))) _Float16 v16h;
typedef __attribute__((ext_vector_type(8)))  float    v8f;
typedef __attribute__((ext_vector_type(2)))  __fp16   h2;
typedef __attribute__((ext_vector_type(4)))  unsigned int u32x4;
typedef __attribute__((ext_vector_type(8)))  unsigned int u32x8;

#define IN_F    64
#define OUT_F   64
#define KN      7
#define KDIM    (KN * IN_F)                  // 448
#define KCHUNKS (KDIM / 32)                  // 14
#define CTILES  (OUT_F / 16)                 // 4
#define B_FRAG_ROWS   (KCHUNKS * CTILES * 32)        // 1792 (lane-rows of 16 halves)
#define B_LDS_HALVES  (B_FRAG_ROWS * 16)             // 28672 halves = 57344 bytes
#define B_BYTES       (B_LDS_HALVES * 2)             // 57344
#define B_QWORDS      (B_BYTES / 8)                  // 7168 (8-byte TDM elements)

// ---------------------------------------------------------------------------
// Pass 1: convert W (f32 [64 x 448] row-major) into pre-swizzled f16 B
// fragments in workspace. Fragment row g = (c, ct, lane); 16 consecutive
// halves per row:  element j -> W[n*KDIM + k],
//   n = ct*16 + (lane & 15),  k = c*32 + (lane>>4)*16 + j
// ---------------------------------------------------------------------------
__global__ __launch_bounds__(256)
void swizzle_W_f16(const float* __restrict__ W, _Float16* __restrict__ wsw) {
    const int g = blockIdx.x * 256 + threadIdx.x;        // 0..1791
    if (g >= B_FRAG_ROWS) return;
    const int lane = g & 31;
    const int ct   = (g >> 5) & 3;
    const int c    = g >> 7;
    const int k0 = c * 32 + (lane >> 4) * 16;
    const int n  = ct * 16 + (lane & 15);

    const float4* src = (const float4*)(W + n * KDIM + k0);  // 64B aligned
    const float4 f0 = src[0], f1 = src[1], f2 = src[2], f3 = src[3];

    union { v16h v; h2 p[8]; } t;
    t.p[0] = __builtin_amdgcn_cvt_pkrtz(f0.x, f0.y);
    t.p[1] = __builtin_amdgcn_cvt_pkrtz(f0.z, f0.w);
    t.p[2] = __builtin_amdgcn_cvt_pkrtz(f1.x, f1.y);
    t.p[3] = __builtin_amdgcn_cvt_pkrtz(f1.z, f1.w);
    t.p[4] = __builtin_amdgcn_cvt_pkrtz(f2.x, f2.y);
    t.p[5] = __builtin_amdgcn_cvt_pkrtz(f2.z, f2.w);
    t.p[6] = __builtin_amdgcn_cvt_pkrtz(f3.x, f3.y);
    t.p[7] = __builtin_amdgcn_cvt_pkrtz(f3.z, f3.w);
    ((v16h*)wsw)[g] = t.v;
}

// ---------------------------------------------------------------------------
// Pass 2: gathered GEMM.  Each wave owns a 16-row x 64-col output strip.
// W fragments are DMA'd into LDS once per block by the Tensor Data Mover.
// ---------------------------------------------------------------------------
__global__ __launch_bounds__(256)
void spherical_conv_wmma(const float*    __restrict__ x,
                         const int*      __restrict__ neigh,
                         const _Float16* __restrict__ wsw,
                         const float*    __restrict__ bias,
                         float*          __restrict__ out,
                         int nvert) {
    __shared__ _Float16 Blds[B_LDS_HALVES];

    // Flat (generic) address of the LDS tile: low 32 bits are the LDS byte
    // offset the TDM descriptor wants.  Feeding this through the asm operand
    // also *escapes* Blds, so the "memory" clobber forces the compiler to
    // treat the DMA as writing it (otherwise the LDS alloc gets DCE'd).
    _Float16* blds_p = Blds;
    const unsigned long long lds_flat = (unsigned long long)(uintptr_t)blds_p;
    const unsigned int lds_off = (unsigned int)lds_flat;

    // ---- TDM: one descriptor copies all 57344 bytes of swizzled W to LDS.
    if (threadIdx.x < 32) {                    // wave 0 only (wave32)
        const unsigned long long ga = (unsigned long long)(uintptr_t)wsw;
        u32x4 g0;
        g0[0] = 1u;                                            // count=1, load
        g0[1] = lds_off;                                       // lds_addr
        g0[2] = (unsigned int)ga;                              // global_addr lo
        g0[3] = (unsigned int)((ga >> 32) & 0x1FFFFFFu)        // global_addr hi
              | (2u << 30);                                    // type = 2
        u32x8 g1;
        g1[0] = 3u << 16;                                      // data_size=8B
        g1[1] = (B_QWORDS & 0xFFFFu) << 16;                    // tensor_dim0 lo
        g1[2] = (B_QWORDS >> 16) | (1u << 16);                 // dim0 hi, tensor_dim1=1
        g1[3] = (B_QWORDS & 0xFFFFu) << 16;                    // tile_dim0
        g1[4] = 0u;                                            // tile_dim1/2 unused
        g1[5] = B_QWORDS;                                      // tensor_dim0_stride lo
        g1[6] = 0u;
        g1[7] = 0u;
        asm volatile("tensor_load_to_lds %0, %1"
                     :: "s"(g0), "s"(g1), "s"(lds_flat) : "memory");
        __builtin_amdgcn_s_wait_tensorcnt(0);
    }
    __syncthreads();

    const int tid  = threadIdx.x;
    const int wave = tid >> 5;
    const int lane = tid & 31;
    const int tile = blockIdx.x * 8 + wave;        // 16-row output tile

    // Clamp tail rows so EXEC stays all-ones through WMMA; mask at store.
    long row = (long)tile * 16 + (lane & 15);
    const int rowc = (row < nvert) ? (int)row : (nvert - 1);

    long nbase[KN];
#pragma unroll
    for (int nb = 0; nb < KN; ++nb)
        nbase[nb] = (long)neigh[rowc * KN + nb] * IN_F;

    v8f acc0 = {}, acc1 = {}, acc2 = {}, acc3 = {};
    const int kg = (lane >> 4) * 16;              // 0 or 16: lane's K sub-group
    const _Float16* __restrict__ blds = blds_p;

#pragma unroll
    for (int c = 0; c < KCHUNKS; ++c) {
        // K-chunk c lives entirely in neighbor (c>>1), feature base (c&1)*32.
        const float* src = x + nbase[c >> 1] + (c & 1) * 32;

        // A fragment (contiguous layout): a[j] -> K = kg + j
        const float4 f0 = *(const float4*)(src + kg);
        const float4 f1 = *(const float4*)(src + kg + 4);
        const float4 f2 = *(const float4*)(src + kg + 8);
        const float4 f3 = *(const float4*)(src + kg + 12);
        union { v16h v; h2 p[8]; } a;
        a.p[0] = __builtin_amdgcn_cvt_pkrtz(f0.x, f0.y);
        a.p[1] = __builtin_amdgcn_cvt_pkrtz(f0.z, f0.w);
        a.p[2] = __builtin_amdgcn_cvt_pkrtz(f1.x, f1.y);
        a.p[3] = __builtin_amdgcn_cvt_pkrtz(f1.z, f1.w);
        a.p[4] = __builtin_amdgcn_cvt_pkrtz(f2.x, f2.y);
        a.p[5] = __builtin_amdgcn_cvt_pkrtz(f2.z, f2.w);
        a.p[6] = __builtin_amdgcn_cvt_pkrtz(f3.x, f3.y);
        a.p[7] = __builtin_amdgcn_cvt_pkrtz(f3.z, f3.w);

        // B fragments: one contiguous 32B LDS read per lane per col-tile.
        const v16h b0 = *(const v16h*)&blds[((c * CTILES + 0) * 32 + lane) * 16];
        const v16h b1 = *(const v16h*)&blds[((c * CTILES + 1) * 32 + lane) * 16];
        const v16h b2 = *(const v16h*)&blds[((c * CTILES + 2) * 32 + lane) * 16];
        const v16h b3 = *(const v16h*)&blds[((c * CTILES + 3) * 32 + lane) * 16];

        acc0 = __builtin_amdgcn_wmma_f32_16x16x32_f16(false, a.v, false, b0, (short)0, acc0, false, false);
        acc1 = __builtin_amdgcn_wmma_f32_16x16x32_f16(false, a.v, false, b1, (short)0, acc1, false, false);
        acc2 = __builtin_amdgcn_wmma_f32_16x16x32_f16(false, a.v, false, b2, (short)0, acc2, false, false);
        acc3 = __builtin_amdgcn_wmma_f32_16x16x32_f16(false, a.v, false, b3, (short)0, acc3, false, false);
    }

    // C/D layout: VGPR r holds M = r + 8*(lane>>4), N = lane&15.
    const int   col0 = lane & 15;
    const int   mlo  = (lane >> 4) * 8;
    const float bb0 = bias[col0];
    const float bb1 = bias[16 + col0];
    const float bb2 = bias[32 + col0];
    const float bb3 = bias[48 + col0];

#pragma unroll
    for (int r = 0; r < 8; ++r) {
        const long orow = (long)tile * 16 + mlo + r;
        if (orow < nvert) {
            float* o = out + orow * OUT_F;
            o[     col0] = acc0[r] + bb0;
            o[16 + col0] = acc1[r] + bb1;
            o[32 + col0] = acc2[r] + bb2;
            o[48 + col0] = acc3[r] + bb3;
        }
    }
}

extern "C" void kernel_launch(void* const* d_in, const int* in_sizes, int n_in,
                              void* d_out, int out_size, void* d_ws, size_t ws_size,
                              hipStream_t stream) {
    const float* x     = (const float*)d_in[0];
    const int*   neigh = (const int*)  d_in[1];
    const float* W     = (const float*)d_in[2];
    const float* bias  = (const float*)d_in[3];
    float*       out   = (float*)d_out;
    _Float16*    wsw   = (_Float16*)d_ws;          // 57344 bytes of scratch

    const int nvert  = in_sizes[0] / IN_F;         // 163842
    const int ntiles = (nvert + 15) / 16;
    const int nblk   = (ntiles + 7) / 8;

    swizzle_W_f16<<<(B_FRAG_ROWS + 255) / 256, 256, 0, stream>>>(W, wsw);
    spherical_conv_wmma<<<nblk, 256, 0, stream>>>(x, neigh, wsw, bias, out, nvert);
}
